// Wave2D_74844100100192
// MI455X (gfx1250) — matrix operand order repaired and checked by reading, not compiled, not run on hardware
//
#include <hip/hip_runtime.h>
#include <hip/hip_bf16.h>
#include <math.h>

typedef __attribute__((ext_vector_type(16))) _Float16 v16h;
typedef __attribute__((ext_vector_type(2)))  __fp16   v2fp;
typedef __attribute__((ext_vector_type(8)))  float    v8f;

#define BD 16
#define CD 384
#define HD 56
#define WD 56
#define HW (HD*WD)          // 3136
#define MD (BD*HW)          // 50176

// ---------------- CDNA5 async global->LDS copy (ASYNCcnt path) ----------------
// GLOBAL_LOAD_ASYNC_TO_LDS_B128: per-lane 16B, LDS byte address in VDST VGPR,
// 64-bit global address in VGPR pair, no SADDR ("off"). Tracked by ASYNCcnt.

__device__ __forceinline__ void async_ld_b128(void* lds_ptr, const void* gptr) {
  uint32_t l = (uint32_t)(uintptr_t)lds_ptr;   // low 32 bits of generic addr = LDS offset
  uint64_t g = (uint64_t)(uintptr_t)gptr;
  asm volatile("global_load_async_to_lds_b128 %0, %1, off" :: "v"(l), "v"(g) : "memory");
}
__device__ __forceinline__ void wait_async0() {
  asm volatile("s_wait_asynccnt 0" ::: "memory");
}

// ---------------- WMMA fragment helpers (A/B in contiguous k order, wave32) ----------------

// A-matrix 16x32 f16: lane L -> row L&15; lanes 0-15 hold K 0..15,
// lanes 16-31 hold K 16..31; 2 halves per VGPR.
__device__ __forceinline__ v16h frag_a(const _Float16* s, int lda, int row0, int k0, int lane) {
  const _Float16* p = s + (row0 + (lane & 15)) * lda + k0 + ((lane >> 4) << 4);
  v16h a;
#pragma unroll
  for (int i = 0; i < 8; ++i) a[i] = p[i];
#pragma unroll
  for (int i = 0; i < 8; ++i) a[8 + i] = p[8 + i];
  return a;
}

// B-matrix 32x16 f16, stored TRANSPOSED in memory as Bt[n][k]:
// lane L -> col L&15; lanes 0-15 hold K 0..15, lanes 16-31 hold K 16..31.
__device__ __forceinline__ v16h frag_bt(const _Float16* s, int ldk, int col0, int k0, int lane) {
  const _Float16* p = s + (col0 + (lane & 15)) * ldk + k0 + ((lane >> 4) << 4);
  v16h b;
#pragma unroll
  for (int i = 0; i < 16; ++i) b[i] = p[i];
  return b;
}

__device__ __forceinline__ v8f wmma32(v16h a, v16h b, v8f c) {
  return __builtin_amdgcn_wmma_f32_16x16x32_f16(false, a, false, b, (short)0, c, false, false);
}

// C/D 16x16 f32: lane L -> col L&15; VGPR r -> row r + 8*(L>>4).

// ---------------- kernel 1: weight prep ----------------

__global__ void prep_kernel(const float* __restrict__ lin_w, const float* __restrict__ tok_w,
                            const float* __restrict__ out_w,
                            _Float16* __restrict__ LW, _Float16* __restrict__ TW,
                            _Float16* __restrict__ OW, _Float16* __restrict__ DM,
                            _Float16* __restrict__ DT) {
  int idx = blockIdx.x * blockDim.x + threadIdx.x;
  const int n1 = 2 * CD * CD;
  const int n2 = CD * CD;
  if (idx < n1) { LW[idx] = (_Float16)lin_w[idx]; return; }
  idx -= n1;
  if (idx < n2) { TW[idx] = (_Float16)tok_w[idx]; return; }
  idx -= n2;
  if (idx < n2) { OW[idx] = (_Float16)out_w[idx]; return; }
  idx -= n2;
  if (idx < 64 * 64) {
    int n = idx >> 6, h = idx & 63;
    float v = 0.f;
    if (n < HD && h < HD) {
      v = cosf((float)n * ((float)h + 0.5f) * (3.14159265358979323846f / (float)HD)) *
          sqrtf(2.0f / (float)HD);
      if (n == 0) v *= 0.70710678118654752440f;
    }
    DM[n * 64 + h] = (_Float16)v;
    DT[h * 64 + n] = (_Float16)v;
  }
}

// ---------------- kernel 2: depthwise 3x3 conv, NCHW -> [M,C] f16 ----------------

__global__ void dwconv_kernel(const float* __restrict__ x, const float* __restrict__ dw_w,
                              const float* __restrict__ dw_b, _Float16* __restrict__ A1) {
  int idx = blockIdx.x * blockDim.x + threadIdx.x;
  if (idx >= BD * CD * HW) return;
  int w = idx % WD; int t = idx / WD;
  int h = t % HD; t /= HD;
  int c = t % CD; int b = t / CD;
  const float* xp = x + (size_t)((b * CD + c) * HD) * WD;
  const float* wp = dw_w + c * 9;
  float acc = dw_b[c];
#pragma unroll
  for (int ky = 0; ky < 3; ++ky) {
    int hh = h + ky - 1;
    if (hh < 0 || hh >= HD) continue;
#pragma unroll
    for (int kx = 0; kx < 3; ++kx) {
      int ww = w + kx - 1;
      if (ww < 0 || ww >= WD) continue;
      acc += xp[hh * WD + ww] * wp[ky * 3 + kx];
    }
  }
  A1[((size_t)(b * HD + h) * WD + w) * CD + c] = (_Float16)acc;
}

// ---------------- kernels 3/4/7: tiled WMMA GEMM, K = 384 ----------------
// C[M,N] = A[M,384] * Bt[N,384]^T + bias, with per-use epilogue.
// Block tile 128x128, 256 threads = 8 waves (4x2), wave tile 32x64.
// f16 A path: tiles staged with GLOBAL_LOAD_ASYNC_TO_LDS_B128.

enum { EPI_LIN = 0, EPI_TOK = 1, EPI_OUT = 2 };

template <bool AF32, int EPI>
__global__ __launch_bounds__(256)
void gemm_kernel(const _Float16* __restrict__ A16, const float* __restrict__ A32,
                 const _Float16* __restrict__ Bt, const float* __restrict__ bias,
                 _Float16* __restrict__ o16a, _Float16* __restrict__ o16b,
                 float* __restrict__ o32, const float* __restrict__ cscal,
                 const float* __restrict__ ascal) {
  __shared__ _Float16 As[128 * 40];
  __shared__ _Float16 Bs[128 * 40];
  const int tid = threadIdx.x;
  const int lane = tid & 31, wid = tid >> 5;
  const int wm = wid & 3, wn = wid >> 2;
  const int m0 = blockIdx.x * 128, n0 = blockIdx.y * 128;

  v8f zero = {};
  v8f acc[2][4];
#pragma unroll
  for (int i = 0; i < 2; ++i)
#pragma unroll
    for (int j = 0; j < 4; ++j) acc[i][j] = zero;

  const int row = tid >> 1;        // 0..127 (f32 staging path)
  const int seg = (tid & 1) * 16;  // 0 / 16

  for (int kb = 0; kb < CD; kb += 32) {
    if (AF32) {
      // manual staging with f32 -> f16 convert (tok GEMM: A = freq_embed)
      const float* p = A32 + (size_t)(m0 + row) * CD + kb + seg;
#pragma unroll
      for (int i = 0; i < 16; ++i) As[row * 40 + seg + i] = (_Float16)p[i];
      const _Float16* q = Bt + (size_t)(n0 + row) * CD + kb + seg;
#pragma unroll
      for (int i = 0; i < 16; ++i) Bs[row * 40 + seg + i] = q[i];
    } else {
      // async DMA: 128 rows x 64B per tile = 512 x 16B chunks, 2 per thread
#pragma unroll
      for (int u = 0; u < 2; ++u) {
        int id = tid + u * 256;
        int r = id >> 2;             // 0..127
        int ch = (id & 3) * 8;       // half offset within 32-half row
        async_ld_b128(&As[r * 40 + ch], A16 + (size_t)(m0 + r) * CD + kb + ch);
        async_ld_b128(&Bs[r * 40 + ch], Bt + (size_t)(n0 + r) * CD + kb + ch);
      }
      wait_async0();
    }
    __syncthreads();
    v16h af[2], bf[4];
#pragma unroll
    for (int i = 0; i < 2; ++i) af[i] = frag_a(As, 40, 32 * wm + 16 * i, 0, lane);
#pragma unroll
    for (int j = 0; j < 4; ++j) bf[j] = frag_bt(Bs, 40, 64 * wn + 16 * j, 0, lane);
#pragma unroll
    for (int i = 0; i < 2; ++i)
#pragma unroll
      for (int j = 0; j < 4; ++j) acc[i][j] = wmma32(af[i], bf[j], acc[i][j]);
    __syncthreads();
  }

  float cv = 0.f, av = 0.f;
  if (EPI == EPI_TOK) { cv = cscal[0]; av = ascal[0]; }
  const int nlane = lane & 15;
  const int mb = (lane >> 4) * 8;
#pragma unroll
  for (int i = 0; i < 2; ++i)
#pragma unroll
    for (int j = 0; j < 4; ++j) {
      int gn = n0 + 64 * wn + 16 * j + nlane;
      float bv = bias[gn];
#pragma unroll
      for (int r = 0; r < 8; ++r) {
        int gm = m0 + 32 * wm + 16 * i + mb + r;
        float v = acc[i][j][r] + bv;
        if (EPI == EPI_LIN) {
          if (gn < CD) {
            o16a[(size_t)gm * CD + gn] = (_Float16)v;      // xx (DCT input)
          } else {
            float g = v / (1.f + __expf(-v));              // silu gate
            o16b[(size_t)gm * CD + (gn - CD)] = (_Float16)g;
          }
        } else if (EPI == EPI_TOK) {
          float t = 0.5f * v * (1.f + erff(v * 0.70710678118f));  // exact gelu
          float ct = cv * t;
          float s = cosf(ct) + sinf(ct) / (cv + 1e-8f) * (1.f + 0.5f * av);
          o32[(size_t)gm * CD + gn] = s;                   // fused trig scale
        } else {  // EPI_OUT: scatter to NCHW f32
          int b = gm / HW; int rem = gm - b * HW;
          int h = rem / WD; int w = rem - h * WD;
          o32[(((size_t)b * CD + gn) * HD + h) * WD + w] = v;
        }
      }
    }
}

// ---------------- kernel 5: DCT2D -> scale -> IDCT2D via 64x64 WMMA tiles ----------------

__device__ __forceinline__ void mm64(const _Float16* Am, const _Float16* Bm, v8f acc[4],
                                     int wid, int lane) {
  v8f zero = {};
#pragma unroll
  for (int j = 0; j < 4; ++j) acc[j] = zero;
#pragma unroll
  for (int k0 = 0; k0 < 64; k0 += 32) {
    v16h a = frag_a(Am, 72, 16 * wid, k0, lane);
#pragma unroll
    for (int j = 0; j < 4; ++j) {
      v16h b = frag_bt(Bm, 72, 16 * j, k0, lane);
      acc[j] = wmma32(a, b, acc[j]);
    }
  }
}

__device__ __forceinline__ void store_acc(_Float16* dst, const v8f acc[4], int row0, int lane,
                                          bool transpose) {
  int nlane = lane & 15, mb = (lane >> 4) * 8;
  if (transpose) {
    // dst[n][m]: consecutive r are consecutive m -> pack 2xf16 (v_cvt_pk_rtz), b32 stores
#pragma unroll
    for (int j = 0; j < 4; ++j) {
      int n = 16 * j + nlane;
#pragma unroll
      for (int r = 0; r < 8; r += 2) {
        v2fp p = __builtin_amdgcn_cvt_pkrtz(acc[j][r], acc[j][r + 1]);
        *(v2fp*)&dst[n * 72 + row0 + mb + r] = p;   // (row0+mb+r) even -> 4B aligned
      }
    }
  } else {
#pragma unroll
    for (int j = 0; j < 4; ++j)
#pragma unroll
      for (int r = 0; r < 8; ++r)
        dst[(row0 + mb + r) * 72 + 16 * j + nlane] = (_Float16)acc[j][r];
  }
}

__global__ __launch_bounds__(128)
void dct_kernel(const _Float16* __restrict__ XX, const float* __restrict__ S,
                const _Float16* __restrict__ Dg, const _Float16* __restrict__ Dtg,
                float* __restrict__ Y) {
  __shared__ _Float16 Ds[64 * 72], Dts[64 * 72], BX[64 * 72], BT[64 * 72];
  const int tid = threadIdx.x, lane = tid & 31, wid = tid >> 5;
  const int b = blockIdx.x / (CD / 8);
  const int cbase = (blockIdx.x % (CD / 8)) * 8;

  // async DMA the two 64x64 f16 DCT matrices into padded LDS (8 x 16B per row)
  for (int i = tid; i < 512; i += 128) {
    int r = i >> 3, ch = (i & 7) * 8;
    async_ld_b128(&Ds[r * 72 + ch], Dg + r * 64 + ch);
    async_ld_b128(&Dts[r * 72 + ch], Dtg + r * 64 + ch);
  }
  wait_async0();

  const int nlane = lane & 15, mb = (lane >> 4) * 8;
  v8f a4[4];

  for (int cc = 0; cc < 8; ++cc) {
    const int c = cbase + cc;
    __syncthreads();
    // load X transposed: BX[w][h] = xx[b,h,w,c] (zero padded to 64)
    for (int i = tid; i < 64 * 64; i += 128) {
      int w = i >> 6, h = i & 63;
      _Float16 v = (_Float16)0.f;
      if (h < HD && w < WD) v = XX[((size_t)(b * HD + h) * WD + w) * CD + c];
      BX[w * 72 + h] = v;
    }
    __syncthreads();
    // T1 = D * X           (A = D rows, Bt = X^T)
    mm64(Ds, BX, a4, wid, lane);
    __syncthreads();
    store_acc(BT, a4, 16 * wid, lane, false);
    __syncthreads();
    // U = T1 * D^T         (A = T1, Bt = D) ; then elementwise trig scale
    mm64(BT, Ds, a4, wid, lane);
#pragma unroll
    for (int j = 0; j < 4; ++j)
#pragma unroll
      for (int r = 0; r < 8; ++r) {
        int n = 16 * wid + mb + r, m = 16 * j + nlane;
        int nn = n < HD ? n : HD - 1;
        int mm = m < WD ? m : WD - 1;
        a4[j][r] *= S[((size_t)(b * HD + nn) * WD + mm) * CD + c];
      }
    __syncthreads();
    store_acc(BX, a4, 16 * wid, lane, false);
    __syncthreads();
    // F2 = F * D (cols y)  (A = F, Bt = D^T); store transposed for next stage
    mm64(BX, Dts, a4, wid, lane);
    __syncthreads();
    store_acc(BT, a4, 16 * wid, lane, true);
    __syncthreads();
    // Ymat = D^T * F2      (A = D^T, Bt = F2^T)
    mm64(Dts, BT, a4, wid, lane);
#pragma unroll
    for (int j = 0; j < 4; ++j)
#pragma unroll
      for (int r = 0; r < 8; ++r) {
        int xr = 16 * wid + mb + r, yc = 16 * j + nlane;
        if (xr < HD && yc < WD)
          Y[((size_t)(b * HD + xr) * WD + yc) * CD + c] = a4[j][r];
      }
  }
}

// ---------------- kernel 6: LayerNorm(C) * gate -> f16 ----------------

__global__ __launch_bounds__(256)
void ln_kernel(const float* __restrict__ Y, const _Float16* __restrict__ G,
               const float* __restrict__ ln_g, const float* __restrict__ ln_b,
               _Float16* __restrict__ A3) {
  int wid = threadIdx.x >> 5, lane = threadIdx.x & 31;
  int m = blockIdx.x * 8 + wid;
  if (m >= MD) return;
  const float* y = Y + (size_t)m * CD;
  float vals[12];
  float s = 0.f, sq = 0.f;
#pragma unroll
  for (int i = 0; i < 12; ++i) {
    float v = y[lane + 32 * i];
    vals[i] = v; s += v; sq += v * v;
  }
#pragma unroll
  for (int o = 16; o > 0; o >>= 1) {
    s += __shfl_xor(s, o, 32);
    sq += __shfl_xor(sq, o, 32);
  }
  float mu = s * (1.f / CD);
  float var = sq * (1.f / CD) - mu * mu;
  float rstd = rsqrtf(var + 1e-5f);
  const _Float16* g = G + (size_t)m * CD;
#pragma unroll
  for (int i = 0; i < 12; ++i) {
    int ch = lane + 32 * i;
    float yn = (vals[i] - mu) * rstd * ln_g[ch] + ln_b[ch];
    A3[(size_t)m * CD + ch] = (_Float16)(yn * (float)g[ch]);
  }
}

// ---------------- launch ----------------

extern "C" void kernel_launch(void* const* d_in, const int* in_sizes, int n_in,
                              void* d_out, int out_size, void* d_ws, size_t ws_size,
                              hipStream_t stream) {
  const float* x     = (const float*)d_in[0];
  const float* fe    = (const float*)d_in[1];
  const float* dw_w  = (const float*)d_in[2];
  const float* dw_b  = (const float*)d_in[3];
  const float* lin_w = (const float*)d_in[4];
  const float* lin_b = (const float*)d_in[5];
  const float* tok_w = (const float*)d_in[6];
  const float* tok_b = (const float*)d_in[7];
  const float* ln_g  = (const float*)d_in[8];
  const float* ln_b  = (const float*)d_in[9];
  const float* out_w = (const float*)d_in[10];
  const float* out_b = (const float*)d_in[11];
  const float* cs    = (const float*)d_in[12];
  const float* al    = (const float*)d_in[13];
  float* out = (float*)d_out;

  char* ws = (char*)d_ws;
  size_t off = 0;
  auto carve = [&](size_t bytes) {
    char* p = ws + off;
    off += (bytes + 255) & ~(size_t)255;
    return p;
  };
  _Float16* A1   = (_Float16*)carve((size_t)MD * CD * 2);   // conv out [M,C] f16
  _Float16* XXb  = (_Float16*)carve((size_t)MD * CD * 2);   // xx f16
  _Float16* GATE = (_Float16*)carve((size_t)MD * CD * 2);   // silu(z) f16
  float*    SB   = (float*)   carve((size_t)MD * CD * 4);   // trig scale f32
  float*    YB   = (float*)   carve((size_t)MD * CD * 4);   // idct out f32
  _Float16* A3   = (_Float16*)carve((size_t)MD * CD * 2);   // ln*gate f16
  _Float16* LW   = (_Float16*)carve((size_t)2 * CD * CD * 2);
  _Float16* TW   = (_Float16*)carve((size_t)CD * CD * 2);
  _Float16* OW   = (_Float16*)carve((size_t)CD * CD * 2);
  _Float16* DM   = (_Float16*)carve(64 * 64 * 2);
  _Float16* DT   = (_Float16*)carve(64 * 64 * 2);
  (void)ws_size; (void)in_sizes; (void)n_in; (void)out_size;

  const int prepN = 2 * CD * CD + CD * CD + CD * CD + 64 * 64;
  prep_kernel<<<(prepN + 255) / 256, 256, 0, stream>>>(lin_w, tok_w, out_w, LW, TW, OW, DM, DT);
  dwconv_kernel<<<(BD * CD * HW + 255) / 256, 256, 0, stream>>>(x, dw_w, dw_b, A1);
  gemm_kernel<false, EPI_LIN><<<dim3(MD / 128, (2 * CD) / 128), 256, 0, stream>>>(
      A1, nullptr, LW, lin_b, XXb, GATE, nullptr, nullptr, nullptr);
  gemm_kernel<true, EPI_TOK><<<dim3(MD / 128, CD / 128), 256, 0, stream>>>(
      nullptr, fe, TW, tok_b, nullptr, nullptr, SB, cs, al);
  dct_kernel<<<BD * (CD / 8), 128, 0, stream>>>(XXb, SB, DM, DT, YB);
  ln_kernel<<<MD / 8, 256, 0, stream>>>(YB, GATE, ln_g, ln_b, A3);
  gemm_kernel<false, EPI_OUT><<<dim3(MD / 128, CD / 128), 256, 0, stream>>>(
      A3, nullptr, OW, out_b, nullptr, nullptr, out, nullptr, nullptr);
}
